// GAT_68968584839686
// MI455X (gfx1250) — matrix operand order repaired and checked by reading, not compiled, not run on hardware
//
#include <hip/hip_runtime.h>
#include <math.h>

// Problem constants (match reference)
#define NODES   100000
#define EDGES   1600000
#define ETOT    (EDGES + NODES)   // self-loops appended
#define IN_CH   128
#define HID     32
#define HEADS1  2
#define OUT_CH  32
#define NEG_SLOPE 0.2f

typedef __bf16 bf16;
typedef __attribute__((ext_vector_type(16))) __bf16 v16bf;
typedef __attribute__((ext_vector_type(8)))  __bf16 v8bf;
typedef __attribute__((ext_vector_type(8)))  float  v8f;

// ---------------------------------------------------------------------------
// Small helpers
// ---------------------------------------------------------------------------
__device__ __forceinline__ void atomicMaxF(float* addr, float val) {
    // Monotone bit trick: positives via signed max, negatives via unsigned
    // min. Init value must be -inf. Lowers to global_atomic_max/min.
    if (val >= 0.0f) atomicMax((int*)addr, __float_as_int(val));
    else             atomicMin((unsigned int*)addr, __float_as_uint(val));
}

__global__ void fill_f32(float* __restrict__ p, float v, long n) {
    long i = (long)blockIdx.x * blockDim.x + threadIdx.x;
    if (i < n) p[i] = v;
}

__global__ void cvt_f32_bf16(const float* __restrict__ in, bf16* __restrict__ out, long n) {
    long i = (long)blockIdx.x * blockDim.x + threadIdx.x;
    if (i < n) out[i] = (bf16)in[i];
}

// ---------------------------------------------------------------------------
// Pack B (fp32 row-major [K,NC]) into WMMA B-fragment order, bf16:
//   Bp[((ct*(K/32)+ks)*32 + lane)*16 + i] = B[ks*32+(lane>>4)*16+i][ct*16+(lane&15)]
// One thread per (ct,ks,lane); each writes 16 contiguous bf16 (32B).
// In the GEMM each lane then loads its fragment as one aligned v16bf.
// ---------------------------------------------------------------------------
template<int K, int NC>
__global__ void pack_B_bf16(const float* __restrict__ Bsrc, bf16* __restrict__ Bp) {
    const int NT = (NC / 16) * (K / 32) * 32;
    int t = blockIdx.x * blockDim.x + threadIdx.x;
    if (t >= NT) return;
    int lane = t & 31;
    int ks   = (t >> 5) % (K / 32);
    int ct   = (t >> 5) / (K / 32);
    int col  = ct * 16 + (lane & 15);
    int krow = ks * 32 + (lane >> 4) * 16;
    bf16* dst = Bp + (size_t)t * 16;
#pragma unroll
    for (int i = 0; i < 16; ++i)
        dst[i] = (bf16)Bsrc[(size_t)(krow + i) * NC + col];
}

// ---------------------------------------------------------------------------
// WMMA GEMM: C[M,NC] = A[M,K] * Bp(packed)   (A bf16 row-major, C fp32)
//   K % 32 == 0, NC % 16 == 0, M % (16*MT) == 0
//   blockDim.x = 32 * (NC/16); gridDim.x = M / (16*MT)
// VGPR layouts used here (A/B k order contiguous):
//   A 16x32 bf16 : lane&15 = row M; lanes 0-15 hold K {0..15},
//                  lanes 16-31 hold K {16..31}
//   B 32x16 bf16 : lane&15 = col N; lanes 0-15 elem i = K i,
//                  lanes 16-31 elem i = K 16+i   (pre-packed, contiguous)
//   C 16x16 f32  : elem v of lane l = C[(l>>4)*8 + v][l&15]
// ---------------------------------------------------------------------------
template<int K, int NC, int MT>
__global__ void wmma_gemm_bf16(const bf16* __restrict__ A,
                               const bf16* __restrict__ Bp,
                               float* __restrict__ C) {
    const int lane    = threadIdx.x & 31;
    const int wave    = threadIdx.x >> 5;      // column-tile index
    const int colbase = wave * 16;
    const int half    = lane >> 4;             // 0 or 1
    const int l15     = lane & 15;

    // Preload this wave's B fragments: one contiguous v16bf per k-step.
    const v16bf* Bv = (const v16bf*)Bp;
    v16bf bfrag[K / 32];
#pragma unroll
    for (int ks = 0; ks < K / 32; ++ks)
        bfrag[ks] = Bv[(wave * (K / 32) + ks) * 32 + lane];

    const long rowblock = (long)blockIdx.x * (16 * MT);
#pragma unroll
    for (int mt = 0; mt < MT; ++mt) {
        const long row   = rowblock + mt * 16 + l15;   // both halves: same row
        const bf16* arow = A + row * K;
        v8f acc = {};                                   // C starts at 0
#pragma unroll
        for (int ks = 0; ks < K / 32; ++ks) {
            // elems 0..7  -> K = ks*32 + half*16 +     i   (contiguous, 16B)
            // elems 8..15 -> K = ks*32 + half*16 + 8 + i   (contiguous, 16B)
            v8bf lo = *(const v8bf*)(arow + ks * 32 + half * 16);
            v8bf hi = *(const v8bf*)(arow + ks * 32 + half * 16 + 8);
            v16bf av;
#pragma unroll
            for (int i = 0; i < 8; ++i) { av[i] = lo[i]; av[i + 8] = hi[i]; }
            acc = __builtin_amdgcn_wmma_f32_16x16x32_bf16(
                /*neg_a=*/false, av, /*neg_b=*/false, bfrag[ks],
                /*c_mod=*/(short)0, acc, /*reuse_a=*/false, /*reuse_b=*/false);
        }
        float* cbase = C + (rowblock + mt * 16 + half * 8) * (long)NC + colbase + l15;
#pragma unroll
        for (int v = 0; v < 8; ++v)
            cbase[(long)v * NC] = acc[v];
    }
}

// ---------------------------------------------------------------------------
// Per-node attention scores: a_s[n,h] = sum_c h[n,h,c]*att_src[h,c], same a_d
// ---------------------------------------------------------------------------
__global__ void attn_scores(const float* __restrict__ h,
                            const float* __restrict__ att_s,
                            const float* __restrict__ att_d,
                            float* __restrict__ a_s, float* __restrict__ a_d,
                            int n, int H, int C) {
    int i = blockIdx.x * blockDim.x + threadIdx.x;
    if (i >= n) return;
    const float* hr = h + (long)i * H * C;
    for (int hh = 0; hh < H; ++hh) {
        float ss = 0.f, dd = 0.f;
        for (int c = 0; c < C; ++c) {
            float v = hr[hh * C + c];
            ss += v * att_s[hh * C + c];
            dd += v * att_d[hh * C + c];
        }
        a_s[(long)i * H + hh] = ss;
        a_d[(long)i * H + hh] = dd;
    }
}

// ---------------------------------------------------------------------------
// Edge pass 1: alpha = leaky_relu(a_s[src]+a_d[dst]); atomic max into m[dst]
// edge e < EDGES: (ei[e], ei[EDGES+e]); e >= EDGES: self-loop node e-EDGES
// ---------------------------------------------------------------------------
__global__ void edge_max(const int* __restrict__ ei,
                         const float* __restrict__ a_s, const float* __restrict__ a_d,
                         float* __restrict__ m, int H) {
    int e = blockIdx.x * blockDim.x + threadIdx.x;
    if (e >= ETOT) return;
    int s = (e < EDGES) ? ei[e]         : (e - EDGES);
    int d = (e < EDGES) ? ei[EDGES + e] : (e - EDGES);
    for (int hh = 0; hh < H; ++hh) {
        float al = a_s[(long)s * H + hh] + a_d[(long)d * H + hh];
        al = (al > 0.f) ? al : al * NEG_SLOPE;
        atomicMaxF(&m[(long)d * H + hh], al);
    }
}

// Edge pass 2: ex = exp(alpha - m[dst]); store per-edge, atomic sum into denom
__global__ void edge_exp(const int* __restrict__ ei,
                         const float* __restrict__ a_s, const float* __restrict__ a_d,
                         const float* __restrict__ m,
                         float* __restrict__ ex, float* __restrict__ denom, int H) {
    int e = blockIdx.x * blockDim.x + threadIdx.x;
    if (e >= ETOT) return;
    int s = (e < EDGES) ? ei[e]         : (e - EDGES);
    int d = (e < EDGES) ? ei[EDGES + e] : (e - EDGES);
    for (int hh = 0; hh < H; ++hh) {
        float al = a_s[(long)s * H + hh] + a_d[(long)d * H + hh];
        al = (al > 0.f) ? al : al * NEG_SLOPE;
        float v = __expf(al - m[(long)d * H + hh]);
        ex[(long)e * H + hh] = v;
        atomicAdd(&denom[(long)d * H + hh], v);
    }
}

// Edge pass 3: out[dst] += h[src] * (ex/denom[dst]); one wave per edge,
// lanes sweep channels -> coalesced gather + coalesced atomic scatter.
__global__ void edge_scatter(const int* __restrict__ ei,
                             const float* __restrict__ h,
                             const float* __restrict__ ex,
                             const float* __restrict__ denom,
                             float* __restrict__ out, int H, int C) {
    int e    = blockIdx.x * (blockDim.x >> 5) + (threadIdx.x >> 5);
    int lane = threadIdx.x & 31;
    if (e >= ETOT) return;
    int s = (e < EDGES) ? ei[e]         : (e - EDGES);
    int d = (e < EDGES) ? ei[EDGES + e] : (e - EDGES);
    int HC = H * C;
    for (int base = lane; base < HC; base += 32) {
        int hh = base / C;  // wave-uniform per iteration (C == 32)
        float coef = ex[(long)e * H + hh] / denom[(long)d * H + hh];
        atomicAdd(&out[(long)d * HC + base], h[(long)s * HC + base] * coef);
    }
}

// act = relu(in + bias), stored as bf16 (feeds the layer-2 WMMA GEMM)
__global__ void relu_bias_bf16(const float* __restrict__ in,
                               const float* __restrict__ bias,
                               bf16* __restrict__ outb, long n, int HC) {
    long i = (long)blockIdx.x * blockDim.x + threadIdx.x;
    if (i >= n) return;
    float v = in[i] + bias[i % HC];
    outb[i] = (bf16)(v > 0.f ? v : 0.f);
}

__global__ void add_bias(float* __restrict__ out, const float* __restrict__ bias,
                         long n, int C) {
    long i = (long)blockIdx.x * blockDim.x + threadIdx.x;
    if (i < n) out[i] += bias[i % C];
}

// ---------------------------------------------------------------------------
extern "C" void kernel_launch(void* const* d_in, const int* in_sizes, int n_in,
                              void* d_out, int out_size, void* d_ws, size_t ws_size,
                              hipStream_t stream) {
    const float* x        = (const float*)d_in[0];
    const int*   ei       = (const int*)  d_in[1];
    const float* W1       = (const float*)d_in[2];
    const float* att_src1 = (const float*)d_in[3];
    const float* att_dst1 = (const float*)d_in[4];
    const float* b1       = (const float*)d_in[5];
    const float* W2       = (const float*)d_in[6];
    const float* att_src2 = (const float*)d_in[7];
    const float* att_dst2 = (const float*)d_in[8];
    const float* b2       = (const float*)d_in[9];
    float* out = (float*)d_out;                      // [NODES, OUT_CH]

    // Carve scratch
    char* ws = (char*)d_ws;
    auto carve = [&](size_t bytes) -> void* {
        void* p = (void*)ws;
        ws += (bytes + 255) & ~(size_t)255;
        return p;
    };
    bf16*  xb    = (bf16*) carve((size_t)NODES * IN_CH * sizeof(bf16));
    bf16*  W1p   = (bf16*) carve((size_t)IN_CH * HEADS1 * HID * sizeof(bf16));   // packed
    bf16*  W2p   = (bf16*) carve((size_t)HEADS1 * HID * OUT_CH * sizeof(bf16));  // packed
    float* h1    = (float*)carve((size_t)NODES * HEADS1 * HID * sizeof(float));
    float* as1   = (float*)carve((size_t)NODES * HEADS1 * sizeof(float));
    float* ad1   = (float*)carve((size_t)NODES * HEADS1 * sizeof(float));
    float* m1    = (float*)carve((size_t)NODES * HEADS1 * sizeof(float));
    float* den1  = (float*)carve((size_t)NODES * HEADS1 * sizeof(float));
    float* ex1   = (float*)carve((size_t)ETOT  * HEADS1 * sizeof(float));
    float* out1  = (float*)carve((size_t)NODES * HEADS1 * HID * sizeof(float));
    bf16*  act1b = (bf16*) carve((size_t)NODES * HEADS1 * HID * sizeof(bf16));
    float* h2    = (float*)carve((size_t)NODES * OUT_CH * sizeof(float));
    float* as2   = (float*)carve((size_t)NODES * sizeof(float));
    float* ad2   = (float*)carve((size_t)NODES * sizeof(float));
    float* m2    = (float*)carve((size_t)NODES * sizeof(float));
    float* den2  = (float*)carve((size_t)NODES * sizeof(float));
    float* ex2   = (float*)carve((size_t)ETOT  * sizeof(float));
    (void)ws_size; (void)n_in; (void)in_sizes; (void)out_size;

    auto blks = [](long n, int t) { return (unsigned)((n + t - 1) / t); };
    const int T = 256;
    const unsigned edge_blks  = blks(ETOT, T);
    const unsigned scat_blks  = blks(ETOT, T / 32);   // one wave per edge

    // ---- input conversion + weight packing --------------------------------
    cvt_f32_bf16<<<blks((long)NODES * IN_CH, T), T, 0, stream>>>(x, xb, (long)NODES * IN_CH);
    pack_B_bf16<IN_CH, HEADS1 * HID><<<2, T, 0, stream>>>(W1, W1p);   // 512 threads
    pack_B_bf16<HEADS1 * HID, OUT_CH><<<1, 128, 0, stream>>>(W2, W2p); // 128 threads

    // ---- Layer 1 ----------------------------------------------------------
    // h1 = xb @ W1 : M=100000, K=128, NC=64; 3125 blocks x 4 waves, 2 M-tiles
    wmma_gemm_bf16<IN_CH, HEADS1 * HID, 2><<<NODES / 32, 128, 0, stream>>>(xb, W1p, h1);
    attn_scores<<<blks(NODES, T), T, 0, stream>>>(h1, att_src1, att_dst1, as1, ad1,
                                                  NODES, HEADS1, HID);
    fill_f32<<<blks((long)NODES * HEADS1, T), T, 0, stream>>>(m1,  -INFINITY, (long)NODES * HEADS1);
    fill_f32<<<blks((long)NODES * HEADS1, T), T, 0, stream>>>(den1, 0.0f,     (long)NODES * HEADS1);
    fill_f32<<<blks((long)NODES * HEADS1 * HID, T), T, 0, stream>>>(out1, 0.0f, (long)NODES * HEADS1 * HID);
    edge_max    <<<edge_blks, T, 0, stream>>>(ei, as1, ad1, m1, HEADS1);
    edge_exp    <<<edge_blks, T, 0, stream>>>(ei, as1, ad1, m1, ex1, den1, HEADS1);
    edge_scatter<<<scat_blks, T, 0, stream>>>(ei, h1, ex1, den1, out1, HEADS1, HID);
    relu_bias_bf16<<<blks((long)NODES * HEADS1 * HID, T), T, 0, stream>>>(
        out1, b1, act1b, (long)NODES * HEADS1 * HID, HEADS1 * HID);

    // ---- Layer 2 ----------------------------------------------------------
    // h2 = act1 @ W2 : M=100000, K=64, NC=32; 3125 blocks x 2 waves
    wmma_gemm_bf16<HEADS1 * HID, OUT_CH, 2><<<NODES / 32, 64, 0, stream>>>(act1b, W2p, h2);
    attn_scores<<<blks(NODES, T), T, 0, stream>>>(h2, att_src2, att_dst2, as2, ad2,
                                                  NODES, 1, OUT_CH);
    fill_f32<<<blks(NODES, T), T, 0, stream>>>(m2,  -INFINITY, NODES);
    fill_f32<<<blks(NODES, T), T, 0, stream>>>(den2, 0.0f,     NODES);
    fill_f32<<<blks((long)NODES * OUT_CH, T), T, 0, stream>>>(out, 0.0f, (long)NODES * OUT_CH);
    edge_max    <<<edge_blks, T, 0, stream>>>(ei, as2, ad2, m2, 1);
    edge_exp    <<<edge_blks, T, 0, stream>>>(ei, as2, ad2, m2, ex2, den2, 1);
    edge_scatter<<<scat_blks, T, 0, stream>>>(ei, h2, ex2, den2, out, 1, OUT_CH);
    // heads=1, concat=False -> mean over 1 head is identity; add bias
    add_bias<<<blks((long)NODES * OUT_CH, T), T, 0, stream>>>(out, b2, (long)NODES * OUT_CH, OUT_CH);
}
